// MFDKT_66829691125843
// MI455X (gfx1250) — matrix operand order repaired and checked by reading, not compiled, not run on hardware
//
#include <hip/hip_runtime.h>

// ---------------- types ----------------
typedef __attribute__((ext_vector_type(16))) __bf16 v16bf;
typedef __attribute__((ext_vector_type(8)))  __bf16 v8bf;
typedef __attribute__((ext_vector_type(8)))  float  v8f;
typedef __attribute__((ext_vector_type(4)))  float  v4f;
typedef __attribute__((ext_vector_type(2)))  unsigned v2u;

#define B_   256
#define S_   200
#define E_   512
#define H_   512
#define SK_  500
#define G4H  2048   // 4*H
#define KC   1024   // combined K (E + H) for step GEMM; padded K for ext GEMM

// ---------------- workspace layout (bytes) ----------------
#define WC_OFF     ((size_t)0)                          // [2048][1024] bf16 combined [W_ih;W_hh]
#define BIAS_OFF   (WC_OFF     + (size_t)G4H*KC*2)      // [2048] f32  (b_ih + b_hh)
#define MFW_OFF    (BIAS_OFF   + (size_t)G4H*4)         // [512][1024] bf16 padded mf_W
#define SIGEXT_OFF (MFW_OFF    + (size_t)H_*KC*2)       // [S][B][H] bf16 sigmoid(ext)
#define HG_OFF     (SIGEXT_OFF + (size_t)S_*B_*H_*2)    // [B][H] bf16 gated h
#define GATES_OFF  (HG_OFF     + (size_t)B_*H_*2)       // [B][4H] f32
#define CT_OFF     (GATES_OFF  + (size_t)B_*G4H*4)      // [B][H] f32
#define HT_OFF     (CT_OFF     + (size_t)B_*H_*4)       // [B][H] f32

// ---------------- helpers ----------------
__device__ __forceinline__ unsigned fbits(float f) {
  unsigned u; __builtin_memcpy(&u, &f, 4); return u;
}
// pack two fp32 into two bf16 (truncate) with one v_perm_b32
__device__ __forceinline__ unsigned pack2bf(float lo, float hi) {
  return __builtin_amdgcn_perm(fbits(hi), fbits(lo), 0x07060302u);
}
__device__ __forceinline__ __bf16 f2bf(float f) {
  unsigned short s = (unsigned short)(fbits(f) >> 16);
  __bf16 o; __builtin_memcpy(&o, &s, 2); return o;
}
__device__ __forceinline__ float bf2f(__bf16 x) {
  unsigned short s; __builtin_memcpy(&s, &x, 2);
  unsigned u = ((unsigned)s) << 16;
  float f; __builtin_memcpy(&f, &u, 4); return f;
}
__device__ __forceinline__ float sigm(float x) { return 1.0f / (1.0f + __expf(-x)); }

__device__ __forceinline__ v8f wmma_bf16(v16bf a, v16bf b, v8f c) {
  return __builtin_amdgcn_wmma_f32_16x16x32_bf16(false, a, false, b, (short)0, c, false, false);
}

// A fragment from an LDS row (row-major, K contiguous).  Same order as B:
// lanes 0-15 take K 0..15, lanes 16-31 take K 16..31.
__device__ __forceinline__ v16bf load_A_lds(const __bf16* row, int k0, int hi) {
  v8bf lo = *(const v8bf*)(row + k0 + hi * 16);
  v8bf hh = *(const v8bf*)(row + k0 + hi * 16 + 8);
  v16bf r;
#pragma unroll
  for (int i = 0; i < 8; ++i) { r[i] = lo[i]; r[i + 8] = hh[i]; }
  return r;
}

// B fragment: weight stored row-major [N][K]; lane = N col (lane&15),
// half-wave selects contiguous K 0..15 / 16..31 of the k0 chunk (same k order as A).
__device__ __forceinline__ v16bf load_B(const __bf16* W, int nbase, int k0, int lane) {
  const __bf16* p = W + (size_t)(nbase + (lane & 15)) * KC + k0 + ((lane >> 4) << 4);
  v8bf lo = *(const v8bf*)p;
  v8bf hh = *(const v8bf*)(p + 8);
  v16bf r;
#pragma unroll
  for (int i = 0; i < 8; ++i) { r[i] = lo[i]; r[i + 8] = hh[i]; }
  return r;
}

// ---------------- weight prep: fp32 -> bf16, combine, pad ----------------
__global__ void prep_kernel(const float* __restrict__ W_ih, const float* __restrict__ W_hh,
                            const float* __restrict__ b_ih, const float* __restrict__ b_hh,
                            const float* __restrict__ mf_W,
                            unsigned* __restrict__ Wc, float* __restrict__ bias,
                            unsigned* __restrict__ mfW) {
  int idx = blockIdx.x * blockDim.x + threadIdx.x;
  int stride = gridDim.x * blockDim.x;
  // Wc: pack 2 bf16 per dword, [N][K/2] dwords
  for (int i = idx; i < G4H * KC / 2; i += stride) {
    int n = i >> 9, k = (i & 511) * 2;
    float a, b;
    if (k < E_) { a = W_ih[n * E_ + k]; b = W_ih[n * E_ + k + 1]; }
    else        { a = W_hh[n * H_ + (k - E_)]; b = W_hh[n * H_ + (k - E_) + 1]; }
    Wc[i] = pack2bf(a, b);
  }
  for (int i = idx; i < G4H; i += stride) bias[i] = b_ih[i] + b_hh[i];
  for (int i = idx; i < H_ * KC / 2; i += stride) {
    int n = i >> 9, k = (i & 511) * 2;
    float a = 0.0f, b = 0.0f;
    if (k + 1 < SK_) { a = mf_W[n * (2 * SK_) + k]; b = mf_W[n * (2 * SK_) + k + 1]; }
    else if (k >= 512 && k + 1 < 512 + SK_) {
      a = mf_W[n * (2 * SK_) + SK_ + (k - 512)];
      b = mf_W[n * (2 * SK_) + SK_ + (k - 512) + 1];
    }
    mfW[i] = pack2bf(a, b);
  }
}

__global__ void init_kernel(__bf16* __restrict__ hg, float* __restrict__ ct) {
  int idx = blockIdx.x * blockDim.x + threadIdx.x;   // B*H threads
  hg[idx] = f2bf(0.0f);
  ct[idx] = 0.0f;
}

// ---------------- ext GEMM: sig_ext[m = s*B+b][h] = sigmoid(ev @ mf_W.T + mf_b) ----------------
// One block per 16-row M tile; ev tile staged zero-padded in LDS (branch-free float4 gather),
// then 8 waves x 4 N-tiles of bf16 WMMA.
__global__ void ext_gemm_kernel(const int* __restrict__ uid_seq, const int* __restrict__ sid_seq,
                                const float* __restrict__ user_emb, const float* __restrict__ skill_emb,
                                const __bf16* __restrict__ mfW, const float* __restrict__ mf_b,
                                __bf16* __restrict__ sig_ext) {
  __shared__ __bf16 As[16 * KC];                // 32 KB
  const int tid = threadIdx.x;
  const int mbase = blockIdx.x * 16;            // m = s*B + b

  // zero the pad columns [500,512) and [1012,1024) of each row
  for (int i = tid; i < 16 * 12; i += 256) {
    int r = i / 12, j = i % 12;
    int col = (j < 6) ? (500 + j * 2) : (1012 + (j - 6) * 2);
    *(unsigned*)&As[r * KC + col] = 0u;
  }
  // gather + convert: per row 125 float4 of user row then 125 float4 of skill row
  for (int i = tid; i < 16 * 250; i += 256) {
    int r = i / 250, j = i % 250;
    int m = mbase + r;
    int s = m >> 8, b = m & 255;                // B == 256
    const float* src;
    int dstcol;
    if (j < 125) { src = user_emb + (size_t)uid_seq[b * S_ + s] * SK_ + j * 4;        dstcol = j * 4; }
    else { int jj = j - 125; src = skill_emb + (size_t)sid_seq[b * S_ + s] * SK_ + jj * 4; dstcol = 512 + jj * 4; }
    v4f f = *(const v4f*)src;
    v2u p = { pack2bf(f[0], f[1]), pack2bf(f[2], f[3]) };
    *(v2u*)&As[r * KC + dstcol] = p;
  }
  __syncthreads();

  const int lane = tid & 31, wave = tid >> 5;
  const int row = lane & 15, hi = lane >> 4;
  const __bf16* arow = As + row * KC;
  v8f acc[4] = {};
  for (int k0 = 0; k0 < KC; k0 += 32) {
    v16bf a = load_A_lds(arow, k0, hi);
#pragma unroll
    for (int j = 0; j < 4; ++j) {
      v16bf bf = load_B(mfW, (wave * 4 + j) * 16, k0, lane);
      acc[j] = wmma_bf16(a, bf, acc[j]);
    }
  }
  const int ncol = lane & 15;
#pragma unroll
  for (int j = 0; j < 4; ++j) {
    int col = (wave * 4 + j) * 16 + ncol;
    float bv = mf_b[col];
#pragma unroll
    for (int r = 0; r < 8; ++r) {
      int mm = mbase + hi * 8 + r;
      sig_ext[(size_t)mm * H_ + col] = f2bf(sigm(acc[j][r] + bv));
    }
  }
}

// ---------------- per-step GEMM: gates = [x_t | hg] @ Wc.T + bias ----------------
// A tile staged in LDS: x half converted via v_perm packing, h half copied with the
// CDNA5 async global->LDS engine.  Each wave: 1 A fragment -> 8 WMMAs per k chunk.
__global__ void step_gemm_kernel(const int* __restrict__ tok_seq, const float* __restrict__ enc_emb,
                                 const __bf16* __restrict__ hg, const __bf16* __restrict__ Wc,
                                 const float* __restrict__ bias, float* __restrict__ gates, int t) {
  __shared__ __bf16 As[16 * KC];                // 32 KB
  const int tid = threadIdx.x;
  const int mbase = blockIdx.x * 16;

  // x half: 16 rows x 128 float4 -> bf16
  for (int i = tid; i < 16 * 128; i += 256) {
    int r = i >> 7, j = i & 127;
    const float* xrow = enc_emb + (size_t)tok_seq[(mbase + r) * S_ + t] * E_;
    v4f f = *(const v4f*)(xrow + j * 4);
    v2u p = { pack2bf(f[0], f[1]), pack2bf(f[2], f[3]) };
    *(v2u*)&As[r * KC + j * 4] = p;
  }
  // h half: 16 rows x 512 bf16 = 1024 x 16B chunks, async copy to LDS
  for (int i = tid; i < 1024; i += 256) {
    int r = i >> 6, j = i & 63;
    unsigned lds = (unsigned)(uintptr_t)&As[r * KC + 512 + j * 8];
    unsigned long long g = (unsigned long long)(uintptr_t)(hg + (size_t)(mbase + r) * H_ + j * 8);
    asm volatile("global_load_async_to_lds_b128 %0, %1, off" :: "v"(lds), "v"(g) : "memory");
  }
  asm volatile("s_wait_asynccnt 0x0" ::: "memory");
  __syncthreads();

  const int lane = tid & 31, wave = tid >> 5;
  const int row = lane & 15, hi = lane >> 4;
  const int ntbase = blockIdx.y * 64 + wave * 8;          // 8 N-tiles per wave
  const __bf16* arow = As + row * KC;
  v8f acc[8] = {};
  for (int k0 = 0; k0 < KC; k0 += 32) {
    v16bf a = load_A_lds(arow, k0, hi);
#pragma unroll
    for (int j = 0; j < 8; ++j) {
      v16bf bf = load_B(Wc, (ntbase + j) * 16, k0, lane);
      acc[j] = wmma_bf16(a, bf, acc[j]);
    }
  }
  const int ncol = lane & 15;
#pragma unroll
  for (int j = 0; j < 8; ++j) {
    int n0 = (ntbase + j) * 16 + ncol;
    float bv = bias[n0];
#pragma unroll
    for (int r = 0; r < 8; ++r) {
      int mm = mbase + hi * 8 + r;
      gates[(size_t)mm * G4H + n0] = acc[j][r] + bv;
    }
  }
}

// ---------------- elementwise LSTM cell + next-step gating ----------------
__global__ void cell_kernel(const float* __restrict__ gates, const __bf16* __restrict__ sig_ext,
                            float* __restrict__ ct, float* __restrict__ ht,
                            __bf16* __restrict__ hg, int t) {
  int idx = blockIdx.x * blockDim.x + threadIdx.x;   // B*H threads
  int b = idx >> 9, h = idx & (H_ - 1);
  const float* gr = gates + (size_t)b * G4H;
  float gi = sigm(gr[h]);
  float gf = sigm(gr[H_ + h]);
  float gg = tanhf(gr[2 * H_ + h]);
  float go = sigm(gr[3 * H_ + h]);
  float c  = gf * ct[idx] + gi * gg;
  float hn = go * tanhf(c);
  ct[idx] = c;
  ht[idx] = hn;
  if (t + 1 < S_) {
    float se = bf2f(sig_ext[(size_t)(t + 1) * B_ * H_ + idx]);
    hg[idx] = f2bf(hn * se);   // h gated for next step's GEMM (h0 handled by init=0)
  }
}

// ---------------- decoder: only the target column is needed ----------------
__global__ void decoder_kernel(const float* __restrict__ ht, const float* __restrict__ dec_W,
                               const float* __restrict__ dec_b, const int* __restrict__ target,
                               float* __restrict__ out) {
  int bidx = (blockIdx.x * blockDim.x + threadIdx.x) >> 5;  // one wave per batch row
  int lane = threadIdx.x & 31;
  int tgt = target[bidx];
  const float* hr = ht + (size_t)bidx * H_;
  const float* wr = dec_W + (size_t)tgt * H_;
  float acc = 0.0f;
  for (int h = lane; h < H_; h += 32) acc += hr[h] * wr[h];
#pragma unroll
  for (int off = 16; off > 0; off >>= 1) acc += __shfl_xor(acc, off);
  if (lane == 0) out[bidx] = sigm(acc + dec_b[tgt]);
}

// ---------------- launch ----------------
extern "C" void kernel_launch(void* const* d_in, const int* in_sizes, int n_in,
                              void* d_out, int out_size, void* d_ws, size_t ws_size,
                              hipStream_t stream) {
  const int*   tok       = (const int*)d_in[0];
  const int*   target    = (const int*)d_in[1];
  const int*   uid_seq   = (const int*)d_in[2];
  const int*   sid_seq   = (const int*)d_in[3];
  // d_in[4] attempt_sequence: unused by reference
  const float* enc_emb   = (const float*)d_in[5];
  const float* user_emb  = (const float*)d_in[6];
  const float* skill_emb = (const float*)d_in[7];
  const float* mf_W      = (const float*)d_in[8];
  const float* mf_b      = (const float*)d_in[9];
  const float* W_ih      = (const float*)d_in[10];
  const float* W_hh      = (const float*)d_in[11];
  const float* b_ih      = (const float*)d_in[12];
  const float* b_hh      = (const float*)d_in[13];
  const float* dec_W     = (const float*)d_in[14];
  const float* dec_b     = (const float*)d_in[15];
  float* out = (float*)d_out;

  char* ws = (char*)d_ws;
  __bf16* Wc      = (__bf16*)(ws + WC_OFF);
  float*  bias    = (float*)(ws + BIAS_OFF);
  __bf16* mfW     = (__bf16*)(ws + MFW_OFF);
  __bf16* sig_ext = (__bf16*)(ws + SIGEXT_OFF);
  __bf16* hg      = (__bf16*)(ws + HG_OFF);
  float*  gates   = (float*)(ws + GATES_OFF);
  float*  ct      = (float*)(ws + CT_OFF);
  float*  ht      = (float*)(ws + HT_OFF);

  // 1) weight conversion / combination (bf16, N-major K-contiguous for WMMA B)
  prep_kernel<<<1024, 256, 0, stream>>>(W_ih, W_hh, b_ih, b_hh, mf_W,
                                        (unsigned*)Wc, bias, (unsigned*)mfW);
  // 2) zero the recurrent state (hg_0 = h0 * sigmoid(ext_0) = 0)
  init_kernel<<<(B_ * H_) / 256, 256, 0, stream>>>(hg, ct);
  // 3) big parallel GEMM: sigmoid(ext) for all 200 steps (bf16)
  ext_gemm_kernel<<<S_ * B_ / 16, 256, 0, stream>>>(uid_seq, sid_seq, user_emb, skill_emb,
                                                    mfW, mf_b, sig_ext);
  // 4) sequential LSTM: per step one WMMA GEMM + one elementwise cell
  for (int t = 0; t < S_; ++t) {
    dim3 g(B_ / 16, 2);
    step_gemm_kernel<<<g, 256, 0, stream>>>(tok, enc_emb, hg, Wc, bias, gates, t);
    cell_kernel<<<(B_ * H_) / 256, 256, 0, stream>>>(gates, sig_ext, ct, ht, hg, t);
  }
  // 5) decoder: gathered single-column dot + sigmoid
  decoder_kernel<<<B_ / 8, 256, 0, stream>>>(ht, dec_W, dec_b, target, out);
}
